// CrossWinAttention_59863254172379
// MI455X (gfx1250) — matrix operand order repaired and checked by reading, not compiled, not run on hardware
//
#include <hip/hip_runtime.h>
#include <hip/hip_bf16.h>

// CDNA5 / gfx1250, wave32. All matmuls via v_wmma_f32_16x16x32_f16.
typedef _Float16 f16;
typedef __attribute__((ext_vector_type(16))) _Float16 v16h;
typedef __attribute__((ext_vector_type(8)))  _Float16 v8h;
typedef __attribute__((ext_vector_type(4)))  _Float16 v4h;
typedef __attribute__((ext_vector_type(8)))  float    v8f;

union FragU { v16h v; uint4 u[2]; };

// A-matrix 16x32 f16 fragment from row-major LDS [rows][strideH]:
// element i of lane (lm,lh) = base[(row0+lm)*strideH + col0 + 16*(i>>3) + 8*lh + (i&7)]
__device__ __forceinline__ v16h frag_a(const f16* base, int strideH,
                                       int row0, int col0, int lm, int lh) {
  const f16* p = base + (size_t)(row0 + lm) * strideH + col0 + 8 * lh;
  FragU f;
  f.u[0] = *(const uint4*)(p);
  f.u[1] = *(const uint4*)(p + 16);
  return f.v;
}

// B-matrix 32x16 f16 fragment. B[k][n] is stored transposed: base[(n0+n)*strideH + col0 + k]
// element i of lane (lm,lh) = B[16*(i>>3) + 8*lh + (i&7)][lm]  -> two 8-half runs per lane.
__device__ __forceinline__ v16h frag_b(const f16* base, int strideH,
                                       int n0, int col0, int lm, int lh) {
  const f16* p = base + (size_t)(n0 + lm) * strideH + col0 + 8 * lh;
  FragU f;
  f.u[0] = *(const uint4*)(p);
  f.u[1] = *(const uint4*)(p + 16);
  return f.v;
}

__device__ __forceinline__ v8f wmma16(v16h a, v16h b, v8f c) {
  return __builtin_amdgcn_wmma_f32_16x16x32_f16(false, a, false, b, (short)0, c,
                                                false, false);
}

// ---- LDS layout (bytes), total 155648 < 160KB -> 2 WGs per 320KB WGP ----
#define LDS_ST   0        // per-wave LN stage: 8 * 16x128 f16 = 32768
#define LDS_QH   32768    // 384x32 f16 = 24576
#define LDS_KH   57344    // 384x32 f16 = 24576
#define LDS_VT   81920    // 32x384 f16 = 24576 (V transposed)
#define LDS_WT   106496   // 32x128 f16 = 8192  (per-phase projection weight slice, transposed)
#define LDS_WPT  114688   // 128x32 f16 = 8192  (wp head slice, transposed)
#define LDS_OH   122880   // 384x32 f16 = 24576 (attention output)
#define LDS_PB   147456   // 8 * 16x32 f16 = 8192 (per-wave P bounce)
#define LDS_TOTAL 155648

__global__ void init_out_kernel(const float* __restrict__ skip,
                                const float* __restrict__ bp,
                                float* __restrict__ out, int n) {
  int i = blockIdx.x * blockDim.x + threadIdx.x;
  if (i < n) out[i] = skip[i] + bp[i & 127];
}

__global__ __launch_bounds__(256)
void xwin_attn_kernel(const float* __restrict__ q, const float* __restrict__ k,
                      const float* __restrict__ v,
                      const float* __restrict__ lnq_g, const float* __restrict__ lnq_b,
                      const float* __restrict__ lnk_g, const float* __restrict__ lnk_b,
                      const float* __restrict__ lnv_g, const float* __restrict__ lnv_b,
                      const float* __restrict__ wq, const float* __restrict__ bq,
                      const float* __restrict__ wk, const float* __restrict__ bk,
                      const float* __restrict__ wv, const float* __restrict__ bv,
                      const float* __restrict__ wp, float* __restrict__ out) {
  extern __shared__ __align__(16) char smem[];
  f16* ST  = (f16*)(smem + LDS_ST);
  f16* QH  = (f16*)(smem + LDS_QH);
  f16* KH  = (f16*)(smem + LDS_KH);
  f16* VT  = (f16*)(smem + LDS_VT);
  f16* WT  = (f16*)(smem + LDS_WT);
  f16* WPT = (f16*)(smem + LDS_WPT);
  f16* OH  = (f16*)(smem + LDS_OH);
  f16* PB  = (f16*)(smem + LDS_PB);

  const int tid  = threadIdx.x;
  const int wave = tid >> 5, lane = tid & 31;
  const int lm = lane & 15, lh = lane >> 4;
  const int head = blockIdx.x & 3;
  const int l    = (blockIdx.x >> 2) & 63;
  const int b    = blockIdx.x >> 8;
  const size_t lbase = (size_t)l * 64;
  const int b6 = b * 6;

  // wp head slice, transposed: WPT[n][kk] = wp[(head*32+kk)*128 + n]
  for (int i = tid; i < 128 * 32; i += 256) {
    int n = i >> 5, kk = i & 31;
    WPT[i] = (f16)wp[(head * 32 + kk) * 128 + n];
  }

  const float* src[3]   = {q, k, v};
  const float* gA[3]    = {lnq_g, lnk_g, lnv_g};
  const float* bA[3]    = {lnq_b, lnk_b, lnv_b};
  const float* wA[3]    = {wq, wk, wv};
  const float* biasA[3] = {bq, bk, bv};

  f16* stw = ST + wave * (16 * 128);

  // ---------- Phase 1: LayerNorm + Q/K/V projection (WMMA) ----------
  for (int p = 0; p < 3; ++p) {
    __syncthreads();  // previous phase consumers done with WT
    const float* w = wA[p];
    // WT[n][kk] = w[kk*128 + head*32 + n], n<32, kk<128 (B-layout: transposed)
    for (int i = tid; i < 32 * 128; i += 256) {
      int n = i >> 7, kk = i & 127;
      WT[i] = (f16)w[kk * 128 + head * 32 + n];
    }
    __syncthreads();

    const float* x  = src[p];
    const float4 gv  = *(const float4*)(gA[p] + lane * 4);
    const float4 bv4 = *(const float4*)(bA[p] + lane * 4);
    const float* bias = biasA[p];

    for (int mt = wave; mt < 24; mt += 8) {
      int m0 = mt * 16;
      // LayerNorm 16 token rows into per-wave stage (f16)
      for (int rr = 0; rr < 16; ++rr) {
        int t = m0 + rr;  // token = n_view*64 + w
        size_t addr = ((size_t)(b6 + (t >> 6)) * 4096 + lbase + (t & 63)) * 128;
        float4 xv = *(const float4*)(x + addr + lane * 4);
        float s  = xv.x + xv.y + xv.z + xv.w;
        float s2 = xv.x * xv.x + xv.y * xv.y + xv.z * xv.z + xv.w * xv.w;
#pragma unroll
        for (int m = 1; m < 32; m <<= 1) {
          s  += __shfl_xor(s, m, 32);
          s2 += __shfl_xor(s2, m, 32);
        }
        float mu   = s * (1.f / 128.f);
        float var  = s2 * (1.f / 128.f) - mu * mu;
        float rstd = rsqrtf(var + 1e-5f);
        v4h y;
        y[0] = (f16)((xv.x - mu) * rstd * gv.x + bv4.x);
        y[1] = (f16)((xv.y - mu) * rstd * gv.y + bv4.y);
        y[2] = (f16)((xv.z - mu) * rstd * gv.z + bv4.z);
        y[3] = (f16)((xv.w - mu) * rstd * gv.w + bv4.w);
        *(v4h*)(stw + rr * 128 + lane * 4) = y;
      }
      // A fragments reused for both N tiles
      v16h afr[4];
#pragma unroll
      for (int i = 0; i < 4; ++i) afr[i] = frag_a(stw, 128, 0, i * 32, lm, lh);
#pragma unroll
      for (int nt = 0; nt < 2; ++nt) {
        int nb = nt * 16;
        v8f c = {};
#pragma unroll
        for (int i = 0; i < 4; ++i)
          c = wmma16(afr[i], frag_b(WT, 128, nb, i * 32, lm, lh), c);
        float bi = bias[head * 32 + nb + lm];
        if (p < 2) {
          f16* dst = (p == 0) ? QH : KH;
#pragma unroll
          for (int r = 0; r < 8; ++r)
            dst[(m0 + 8 * lh + r) * 32 + nb + lm] = (f16)(c[r] + bi);
        } else {
          v8h tv;
#pragma unroll
          for (int r = 0; r < 8; ++r) tv[r] = (f16)(c[r] + bi);
          *(v8h*)(VT + (size_t)(nb + lm) * 384 + m0 + 8 * lh) = tv;  // transposed
        }
      }
    }
  }
  __syncthreads();

  // ---------- Phase 2: flash attention per wave over 16-query tiles ----------
  const float c2 = 0.17677669529663687f * 1.4426950408889634f;  // 1/sqrt(32) * log2(e)
  f16* Pw = PB + wave * (16 * 32);
  for (int mt = wave; mt < 24; mt += 8) {
    int m0 = mt * 16;
    v16h aq = frag_a(QH, 32, m0, 0, lm, lh);
    float mrun[8], lrun[8];
#pragma unroll
    for (int r = 0; r < 8; ++r) { mrun[r] = -1e30f; lrun[r] = 0.f; }
    v8f o0 = {}, o1 = {};
    for (int kb = 0; kb < 384; kb += 32) {
      v8f z = {};
      v8f s0 = wmma16(aq, frag_b(KH, 32, kb,      0, lm, lh), z);
      v8f s1 = wmma16(aq, frag_b(KH, 32, kb + 16, 0, lm, lh), z);
#pragma unroll
      for (int r = 0; r < 8; ++r) {
        float a0 = s0[r] * c2, a1 = s1[r] * c2;
        float mx = fmaxf(a0, a1);
#pragma unroll
        for (int m = 1; m < 16; m <<= 1) mx = fmaxf(mx, __shfl_xor(mx, m, 32));
        float mn = fmaxf(mrun[r], mx);
        float corr = __builtin_amdgcn_exp2f(mrun[r] - mn);
        mrun[r] = mn;
        o0[r] *= corr; o1[r] *= corr; lrun[r] *= corr;
        float p0 = __builtin_amdgcn_exp2f(a0 - mn);
        float p1 = __builtin_amdgcn_exp2f(a1 - mn);
        float ps = p0 + p1;
#pragma unroll
        for (int m = 1; m < 16; m <<= 1) ps += __shfl_xor(ps, m, 32);
        lrun[r] += ps;
        Pw[(8 * lh + r) * 32 + lm]      = (f16)p0;  // keys kb..kb+15
        Pw[(8 * lh + r) * 32 + 16 + lm] = (f16)p1;  // keys kb+16..kb+31
      }
      // same-wave LDS ops are in-order: safe to re-read Pw as A fragment
      v16h ap = frag_a(Pw, 32, 0, 0, lm, lh);
      o0 = wmma16(ap, frag_b(VT, 384, 0,  kb, lm, lh), o0);
      o1 = wmma16(ap, frag_b(VT, 384, 16, kb, lm, lh), o1);
    }
#pragma unroll
    for (int r = 0; r < 8; ++r) {
      float inv = 1.f / lrun[r];
      OH[(m0 + 8 * lh + r) * 32 + lm]      = (f16)(o0[r] * inv);
      OH[(m0 + 8 * lh + r) * 32 + 16 + lm] = (f16)(o1[r] * inv);
    }
  }
  __syncthreads();

  // ---------- Phase 3: O @ wp_head, scaled 1/6 (view mean), atomic accumulate ----------
  const size_t outBase = ((size_t)b * 64 + l) * 64;
  for (int i = 0; i < 24; ++i) {
    int idx = wave * 24 + i;
    int m0 = (idx >> 3) * 16, nb = (idx & 7) * 16;
    v16h a  = frag_a(OH, 32, m0, 0, lm, lh);
    v16h bw = frag_b(WPT, 32, nb, 0, lm, lh);
    v8f z = {};
    v8f c = wmma16(a, bw, z);
#pragma unroll
    for (int r = 0; r < 8; ++r) {
      int rr = m0 + 8 * lh + r;       // token row: view = rr/64 folds into the mean
      int wtok = rr & 63;
      atomicAdd(out + (outBase + wtok) * 128 + nb + lm, c[r] * (1.f / 6.f));
    }
  }
}

extern "C" void kernel_launch(void* const* d_in, const int* in_sizes, int n_in,
                              void* d_out, int out_size, void* d_ws, size_t ws_size,
                              hipStream_t stream) {
  const float* q     = (const float*)d_in[0];
  const float* k     = (const float*)d_in[1];
  const float* v     = (const float*)d_in[2];
  const float* skip  = (const float*)d_in[3];
  const float* lnq_g = (const float*)d_in[4];
  const float* lnq_b = (const float*)d_in[5];
  const float* lnk_g = (const float*)d_in[6];
  const float* lnk_b = (const float*)d_in[7];
  const float* lnv_g = (const float*)d_in[8];
  const float* lnv_b = (const float*)d_in[9];
  const float* wq    = (const float*)d_in[10];
  const float* bq    = (const float*)d_in[11];
  const float* wk    = (const float*)d_in[12];
  const float* bk    = (const float*)d_in[13];
  const float* wv    = (const float*)d_in[14];
  const float* bv    = (const float*)d_in[15];
  const float* wp    = (const float*)d_in[16];
  const float* bp    = (const float*)d_in[17];
  float* out = (float*)d_out;

  init_out_kernel<<<(out_size + 255) / 256, 256, 0, stream>>>(skip, bp, out, out_size);
  // grid = B * L * HEADS = 2 * 64 * 4 = 512 workgroups of 8 waves
  xwin_attn_kernel<<<512, 256, LDS_TOTAL, stream>>>(
      q, k, v, lnq_g, lnq_b, lnk_g, lnk_b, lnv_g, lnv_b,
      wq, bq, wk, bk, wv, bv, wp, out);
}
